// MultiHeadDebiasedAttention_83872121356655
// MI455X (gfx1250) — matrix operand order repaired and checked by reading, not compiled, not run on hardware
//
#include <hip/hip_runtime.h>

// Problem constants (match reference)
#define B_  4
#define S_  2048
#define H_  1024
#define NH_ 16
#define HD_ 64

typedef __attribute__((ext_vector_type(16))) __bf16 v16bf;
typedef __attribute__((ext_vector_type(8)))  float  v8f;
typedef unsigned int v4u __attribute__((ext_vector_type(4)));
typedef int          v8i __attribute__((ext_vector_type(8)));
typedef int          v4i __attribute__((ext_vector_type(4)));

// ---------------------------------------------------------------------------
// TDM helper: issue a tensor_load_to_lds of a 16-row x 64-col bf16 tile that
// starts at global address `gsrc`, into workgroup-LDS byte offset `lds_off`.
// D# packing per CDNA5 ISA sec. 8.3/8.4 (2-D tensor, groups 2/3 zero).
// This toolchain exposes the 6-arg builtin:
//   (uint32x4 g0, int32x8 g1, int32x4, int32x4, int32x8, i32 cpol)
// Completion tracked with TENSORcnt.
// ---------------------------------------------------------------------------
__device__ __forceinline__ void tdm_load_k_tile(const __bf16* gsrc, unsigned lds_off)
{
    const unsigned long long ga = (unsigned long long)(size_t)gsrc;
    v4u g0;
    g0.x = 1u;                                   // count=1, user mode, no gather
    g0.y = lds_off;                              // D#.lds_addr (bytes)
    g0.z = (unsigned)ga;                         // global_addr[31:0]
    g0.w = (unsigned)(ga >> 32) | (2u << 30);    // global_addr[56:32] | type=2
    v8i g1;
    g1[0] = 0x00010000;      // wg_mask=0, data_size=1 (2 bytes/elt)
    g1[1] = (64 << 16);      // tensor_dim0 = 64 elems (bits [63:48])
    g1[2] = (16 << 16);      // tensor_dim1 = 16 rows  (bits [95:80])
    g1[3] = (64 << 16);      // tile_dim0   = 64       (bits [127:112])
    g1[4] = 16;              // tile_dim1   = 16       (bits [143:128])
    g1[5] = 64;              // tensor_dim0_stride = 64 elems (row pitch)
    g1[6] = 0;
    g1[7] = 0;
    v4i z4 = {0, 0, 0, 0};
    v8i z8 = {0, 0, 0, 0, 0, 0, 0, 0};
    __builtin_amdgcn_tensor_load_to_lds(g0, g1, z4, z4, z8, 0);
}

// ---------------------------------------------------------------------------
// Kernel 1: fused QKV projection.  grid = (H/256, B*S/16, 3), block = 128.
// Each wave computes a 16x64 strip (4 WMMA accumulators) so the A fragment is
// loaded once per k-step and reused by 4 v_wmma_f32_16x16x32_bf16 ops.
// Q,K written head-major [b][h][s][d]; V written transposed [b][h][d][s].
// ---------------------------------------------------------------------------
__global__ __launch_bounds__(128)
void qkv_wmma_kernel(const float* __restrict__ X,
                     const float* __restrict__ Wq, const float* __restrict__ bq,
                     const float* __restrict__ Wk, const float* __restrict__ bk,
                     const float* __restrict__ Wv, const float* __restrict__ bv,
                     __bf16* __restrict__ Qd, __bf16* __restrict__ Kd,
                     __bf16* __restrict__ Vd)
{
    const int wave = threadIdx.x >> 5;
    const int lane = threadIdx.x & 31;
    const int hf   = lane >> 4;        // 0: lanes 0-15, 1: lanes 16-31
    const int l16  = lane & 15;

    const int mat = blockIdx.z;        // 0=Q 1=K 2=V
    const float* W  = (mat == 0) ? Wq : (mat == 1) ? Wk : Wv;
    const float* bs = (mat == 0) ? bq : (mat == 1) ? bk : bv;
    __bf16*      dst = (mat == 0) ? Qd : (mat == 1) ? Kd : Vd;

    const int m0     = blockIdx.y * 16;                // row tile over B*S
    const int n_base = blockIdx.x * 256 + wave * 64;   // wave's 64-col strip

    const float* xrow = X + (size_t)(m0 + l16) * H_;   // A: row M = l16
    const float* wrow[4];
    #pragma unroll
    for (int t = 0; t < 4; ++t)
        wrow[t] = W + (size_t)(n_base + t * 16 + l16) * H_;  // B[k][n] = W[n][k]

    v8f acc0 = {}, acc1 = {}, acc2 = {}, acc3 = {};
    for (int k0 = 0; k0 < H_; k0 += 32) {
        v16bf a;
        const float* xa = xrow + k0 + hf * 16;
        #pragma unroll
        for (int e = 0; e < 8; ++e) {          // A 16x32 bf16, contiguous k order (same as B)
            a[e]     = (__bf16)xa[e];          // K = hf*16 + e
            a[e + 8] = (__bf16)xa[8 + e];      // K = hf*16 + 8 + e
        }
        if (k0 + 32 < H_) {
            __builtin_prefetch((const void*)(xrow + k0 + 32), 0, 3);
            __builtin_prefetch((const void*)(wrow[0] + k0 + 32), 0, 3);
        }
        #pragma unroll
        for (int t = 0; t < 4; ++t) {
            v16bf bfrag;                       // B 32x16: lane col n, K = hf*16+e
            const float* wb = wrow[t] + k0 + hf * 16;
            #pragma unroll
            for (int e = 0; e < 16; ++e) bfrag[e] = (__bf16)wb[e];
            v8f& acc = (t == 0) ? acc0 : (t == 1) ? acc1 : (t == 2) ? acc2 : acc3;
            acc = __builtin_amdgcn_wmma_f32_16x16x32_bf16(
                false, a, false, bfrag, (short)0, acc, false, false);
        }
    }

    #pragma unroll
    for (int t = 0; t < 4; ++t) {
        const v8f& acc = (t == 0) ? acc0 : (t == 1) ? acc1 : (t == 2) ? acc2 : acc3;
        const int n  = n_base + t * 16 + l16;
        const float bias = bs[n];
        const int hh = n >> 6;     // head
        const int dd = n & 63;     // dim within head
        #pragma unroll
        for (int r = 0; r < 8; ++r) {          // C/D: M = r + 8*hf, N = l16
            const int m  = m0 + r + 8 * hf;
            const int bb = m >> 11;            // / S_
            const int ss = m & (S_ - 1);
            const __bf16 v = (__bf16)(acc[r] + bias);
            if (mat == 2)   // V transposed [b][h][d][s]
                dst[(((size_t)bb * NH_ + hh) * HD_ + dd) * S_ + ss] = v;
            else            // Q,K head-major [b][h][s][d]
                dst[(((size_t)bb * NH_ + hh) * S_ + ss) * HD_ + dd] = v;
        }
    }
}

// ---------------------------------------------------------------------------
// Kernel 2: attention.  grid = (S/16, NH, B), block = 128 (4 waves).
// LDS: 16x2048 fp32 score strip (128 KB) + 16 scales + per-wave double-
// buffered 2 KB K-tile staging (TDM tensor_load_to_lds, TENSORcnt waits).
// Scores + context via bf16 WMMA; exact fp32 softmax; probs streamed with
// non-temporal stores.  V arrives transposed [b][h][d][s].
// ---------------------------------------------------------------------------
#define SCORE_BYTES ((16 * S_ + 16) * 4)          // scores + rowinv
#define STAGE_BYTES (2048)                        // one 16x64 bf16 tile
#define ATTN_LDS_BYTES (SCORE_BYTES + 4 * 2 * STAGE_BYTES)

__global__ __launch_bounds__(128)
void attn_wmma_kernel(const __bf16* __restrict__ Q, const __bf16* __restrict__ K,
                      const __bf16* __restrict__ Vt,
                      const int* __restrict__ mask,
                      const float* __restrict__ bias_corr,
                      float* __restrict__ probs_out,
                      __bf16* __restrict__ ctx_out)
{
    extern __shared__ float smem[];            // dynamic LDS starts at offset 0
    float* sc     = smem;
    float* rowinv = smem + 16 * S_;

    const int wave = threadIdx.x >> 5;
    const int lane = threadIdx.x & 31;
    const int hf   = lane >> 4;
    const int l16  = lane & 15;

    const int q0 = blockIdx.x * 16;
    const int h  = blockIdx.y;
    const int b  = blockIdx.z;

    const __bf16* Qh  = Q  + (((size_t)b * NH_ + h) * S_) * HD_;
    const __bf16* Kh  = K  + (((size_t)b * NH_ + h) * S_) * HD_;
    const __bf16* Vth = Vt + (((size_t)b * NH_ + h) * HD_) * S_;

    // ---- Q fragments for this 16-row strip (HD=64 -> two K=32 chunks) ----
    v16bf qa0, qa1;
    {
        const __bf16* qrow = Qh + (size_t)(q0 + l16) * HD_;
        #pragma unroll
        for (int e = 0; e < 8; ++e) {
            qa0[e]     = qrow[hf * 16 + e];
            qa0[e + 8] = qrow[hf * 16 + 8 + e];
            qa1[e]     = qrow[32 + hf * 16 + e];
            qa1[e + 8] = qrow[32 + hf * 16 + 8 + e];
        }
    }
    const float bc    = bias_corr[h];
    const float scale = 0.125f;                // 1/sqrt(HD=64)

    // ---- scores: each wave owns k-tiles wave, wave+4, ...  K tiles are
    //      DMA'd into LDS by the Tensor Data Mover, double buffered. ----
    const unsigned my_stage = (unsigned)SCORE_BYTES + (unsigned)wave * 2 * STAGE_BYTES;
    tdm_load_k_tile(Kh + (size_t)(wave * 16) * HD_, my_stage);   // prime buf 0

    int buf = 0;
    for (int kt = wave; kt < S_ / 16; kt += 4) {
        const int ktn = kt + 4;
        if (ktn < S_ / 16) {   // issue next tile into the other buffer
            tdm_load_k_tile(Kh + (size_t)(ktn * 16) * HD_,
                            my_stage + (unsigned)((buf ^ 1) * STAGE_BYTES));
            __builtin_amdgcn_s_wait_tensorcnt(1);   // current buffer ready
        } else {
            __builtin_amdgcn_s_wait_tensorcnt(0);
        }

        const __bf16* kst =
            (const __bf16*)((const char*)smem + my_stage + buf * STAGE_BYTES);

        v16bf b0, b1;                          // B[kk][n] = K[kv0+n][d0+kk]
        const __bf16* krow = kst + l16 * HD_;  // staged row-major 16x64
        #pragma unroll
        for (int e = 0; e < 16; ++e) {
            b0[e] = krow[hf * 16 + e];         // d0 = 0
            b1[e] = krow[32 + hf * 16 + e];    // d0 = 32
        }
        v8f acc = {};
        acc = __builtin_amdgcn_wmma_f32_16x16x32_bf16(
            false, qa0, false, b0, (short)0, acc, false, false);
        acc = __builtin_amdgcn_wmma_f32_16x16x32_bf16(
            false, qa1, false, b1, (short)0, acc, false, false);

        const int kv0 = kt * 16;
        const int mk = mask[b * S_ + kv0 + l16];
        #pragma unroll
        for (int r = 0; r < 8; ++r) {
            float v = acc[r] * scale - bc;
            if (mk == 0) v = -1.0e9f;
            sc[(r + 8 * hf) * S_ + kv0 + l16] = v;
        }
        buf ^= 1;
    }
    __syncthreads();

    // ---- exact softmax over 2048 cols: 8 lanes per row ----
    {
        const int row = threadIdx.x >> 3;
        const int sub = threadIdx.x & 7;
        float* srow = sc + row * S_;
        float mx = -3.4e38f;
        for (int c = sub; c < S_; c += 8) mx = fmaxf(mx, srow[c]);
        #pragma unroll
        for (int off = 1; off < 8; off <<= 1) mx = fmaxf(mx, __shfl_xor(mx, off, 32));
        float sum = 0.0f;
        for (int c = sub; c < S_; c += 8) {
            float e = __expf(srow[c] - mx);
            srow[c] = e;
            sum += e;
        }
        #pragma unroll
        for (int off = 1; off < 8; off <<= 1) sum += __shfl_xor(sum, off, 32);
        if (sub == 0) rowinv[row] = 1.0f / sum;
    }
    __syncthreads();

    // ---- normalize in LDS + coalesced streaming 1 GB probs write ----
    {
        float* pout = probs_out + (((size_t)b * NH_ + h) * S_ + q0) * S_;
        for (int idx = threadIdx.x; idx < 16 * S_; idx += 128) {
            const int rr = idx >> 11;          // S_ == 2048
            float p = sc[idx] * rowinv[rr];
            sc[idx] = p;
            __builtin_nontemporal_store(p, pout + idx);  // NT: bypass L2 residency
        }
    }
    __syncthreads();

    // ---- context: C(16x64) = P(16x2048) x V(2048x64), wave w owns 16 cols ----
    v8f cacc = {};
    const int nd = wave * 16 + l16;            // d column of this lane
    const __bf16* vtrow = Vth + (size_t)nd * S_;   // V^T row: contiguous in kv
    for (int k0 = 0; k0 < S_; k0 += 32) {
        v16bf pa, vb;
        const float* prow = sc + l16 * S_ + k0;
        #pragma unroll
        for (int e = 0; e < 8; ++e) {
            pa[e]     = (__bf16)prow[hf * 16 + e];
            pa[e + 8] = (__bf16)prow[hf * 16 + 8 + e];
        }
        const __bf16* vt = vtrow + k0 + hf * 16;   // B[kk][n]=V[k0+kk][nd]
        #pragma unroll
        for (int e = 0; e < 16; ++e) vb[e] = vt[e];  // contiguous 32 bytes
        cacc = __builtin_amdgcn_wmma_f32_16x16x32_bf16(
            false, pa, false, vb, (short)0, cacc, false, false);
    }
    #pragma unroll
    for (int r = 0; r < 8; ++r) {              // ctx in [b][s][h*HD+d] bf16
        const int s = q0 + r + 8 * hf;
        ctx_out[((size_t)b * S_ + s) * H_ + h * HD_ + nd] = (__bf16)cacc[r];
    }
}

// ---------------------------------------------------------------------------
// Kernel 3: output projection.  grid = (H/256, B*S/16), block = 128.
// out = ctx @ Wo^T + bo, fp32 result.  16x64 strip per wave (4 accumulators).
// ---------------------------------------------------------------------------
__global__ __launch_bounds__(128)
void outproj_wmma_kernel(const __bf16* __restrict__ Ctx,
                         const float* __restrict__ Wo,
                         const float* __restrict__ bo,
                         float* __restrict__ Out)
{
    const int wave = threadIdx.x >> 5;
    const int lane = threadIdx.x & 31;
    const int hf   = lane >> 4;
    const int l16  = lane & 15;

    const int m0     = blockIdx.y * 16;
    const int n_base = blockIdx.x * 256 + wave * 64;

    const __bf16* arow = Ctx + (size_t)(m0 + l16) * H_;
    const float* wrow[4];
    #pragma unroll
    for (int t = 0; t < 4; ++t)
        wrow[t] = Wo + (size_t)(n_base + t * 16 + l16) * H_;

    v8f acc0 = {}, acc1 = {}, acc2 = {}, acc3 = {};
    for (int k0 = 0; k0 < H_; k0 += 32) {
        v16bf a;
        const __bf16* xa = arow + k0 + hf * 16;
        #pragma unroll
        for (int e = 0; e < 8; ++e) {
            a[e]     = xa[e];
            a[e + 8] = xa[8 + e];
        }
        if (k0 + 32 < H_) {
            __builtin_prefetch((const void*)(arow + k0 + 32), 0, 3);
            __builtin_prefetch((const void*)(wrow[0] + k0 + 32), 0, 3);
        }
        #pragma unroll
        for (int t = 0; t < 4; ++t) {
            v16bf bfrag;
            const float* wb = wrow[t] + k0 + hf * 16;
            #pragma unroll
            for (int e = 0; e < 16; ++e) bfrag[e] = (__bf16)wb[e];
            v8f& acc = (t == 0) ? acc0 : (t == 1) ? acc1 : (t == 2) ? acc2 : acc3;
            acc = __builtin_amdgcn_wmma_f32_16x16x32_bf16(
                false, a, false, bfrag, (short)0, acc, false, false);
        }
    }

    #pragma unroll
    for (int t = 0; t < 4; ++t) {
        const v8f& acc = (t == 0) ? acc0 : (t == 1) ? acc1 : (t == 2) ? acc2 : acc3;
        const int n = n_base + t * 16 + l16;
        const float bias = bo[n];
        #pragma unroll
        for (int r = 0; r < 8; ++r) {
            const int m = m0 + r + 8 * hf;
            Out[(size_t)m * H_ + n] = acc[r] + bias;
        }
    }
}

// ---------------------------------------------------------------------------
// Launch.  Workspace layout (bf16):
//   [0,16MB)   Q  [b][h][s][d]
//   [16,32MB)  K  [b][h][s][d]
//   [32,48MB)  V^T [b][h][d][s]
//   [48,64MB)  context [b][s][h]
// d_out: output (B*S*H f32) then attention_probs (B*NH*S*S f32).
// ---------------------------------------------------------------------------
extern "C" void kernel_launch(void* const* d_in, const int* in_sizes, int n_in,
                              void* d_out, int out_size, void* d_ws, size_t ws_size,
                              hipStream_t stream) {
    const float* hidden = (const float*)d_in[0];
    const int*   mask   = (const int*)d_in[1];
    const float* Wq = (const float*)d_in[2];  const float* bq = (const float*)d_in[3];
    const float* Wk = (const float*)d_in[4];  const float* bk = (const float*)d_in[5];
    const float* Wv = (const float*)d_in[6];  const float* bv = (const float*)d_in[7];
    const float* Wo = (const float*)d_in[8];  const float* bo = (const float*)d_in[9];
    const float* bc = (const float*)d_in[10];

    float* out   = (float*)d_out;
    float* probs = out + (size_t)B_ * S_ * H_;

    char* ws = (char*)d_ws;
    const size_t QKV_BYTES = (size_t)B_ * NH_ * S_ * HD_ * 2;  // 16 MB each
    __bf16* Qd = (__bf16*)(ws);
    __bf16* Kd = (__bf16*)(ws + QKV_BYTES);
    __bf16* Vd = (__bf16*)(ws + 2 * QKV_BYTES);
    __bf16* Cd = (__bf16*)(ws + 3 * QKV_BYTES);

    dim3 blk(128, 1, 1);

    qkv_wmma_kernel<<<dim3(H_ / 256, (B_ * S_) / 16, 3), blk, 0, stream>>>(
        hidden, Wq, bq, Wk, bk, Wv, bv, Qd, Kd, Vd);

    attn_wmma_kernel<<<dim3(S_ / 16, NH_, B_), blk, (size_t)ATTN_LDS_BYTES, stream>>>(
        Qd, Kd, Vd, mask, bc, probs, Cd);

    outproj_wmma_kernel<<<dim3(H_ / 256, (B_ * S_) / 16, 1), blk, 0, stream>>>(
        Cd, Wo, bo, out);
}
